// SingleSelfHeadAttention_88235808129314
// MI455X (gfx1250) — matrix operand order repaired and checked by reading, not compiled, not run on hardware
//
#include <hip/hip_runtime.h>
#include <hip/hip_bf16.h>

typedef __attribute__((ext_vector_type(16))) _Float16 v16h;
typedef __attribute__((ext_vector_type(8)))  _Float16 v8h;
typedef __attribute__((ext_vector_type(8)))  float    v8f;

static constexpr int Bn = 4;        // batch
static constexpr int Sq = 2048;     // sequence length
static constexpr int Dm = 1024;     // embed dim
static constexpr int NT = Bn * Sq;  // 8192 flattened tokens

#define DEV static __device__ __forceinline__

// ---- WMMA fragment helpers (wave32, 16x16x32 f16 -> f32) -------------------

// A-matrix 16x32 f16, row-major source with leading dim `ld` (halves).
// Layout: lane L: M = L%16, g = L/16; halves 0..15 -> K = g*16 + 0..15
// (two adjacent 16-byte chunks, matching ld_fragB).
DEV v16h ld_fragA(const _Float16* base, int ld) {
  int lane = threadIdx.x & 31;
  int r = lane & 15;
  int g = lane >> 4;
  const _Float16* p = base + (size_t)r * ld + g * 16;
  v8h lo = *(const v8h*)(p);
  v8h hi = *(const v8h*)(p + 8);
  v16h f;
#pragma unroll
  for (int i = 0; i < 8; ++i) { f[i] = lo[i]; f[i + 8] = hi[i]; }
  return f;
}

// B-matrix 32x16 f16 loaded from the TRANSPOSED operand Bt[N x K] row-major:
// lane L: N = L%16, halves h -> K = (L/16)*16 + h (one contiguous 32-byte run).
DEV v16h ld_fragB(const _Float16* base, int ld) {
  int lane = threadIdx.x & 31;
  int n = lane & 15;
  int g = lane >> 4;
  return *(const v16h*)(base + (size_t)n * ld + g * 16);
}

DEV v8f wmma_f16(v16h a, v16h b, v8f c) {
  return __builtin_amdgcn_wmma_f32_16x16x32_f16(
      /*neg_a=*/false, a, /*neg_b=*/false, b,
      /*c_mod=*/(short)0, c, /*reuse_a=*/false, /*reuse_b=*/false);
}

// ---- 32x64 wave-tile GEMM core with register double-buffering --------------
// Computes acc[2][4] (+=) over nIter k-steps of 32. A row-major [.. x lda],
// Bt row-major [.. x ldb] (transposed B). Loads for step k+1 are issued
// before the WMMAs consuming step k, so vmem latency overlaps matrix math.

struct Frags {
  v16h a[2];
  v16h b[4];
};

DEV void load_frags(Frags& f, const _Float16* ap, const _Float16* bp,
                    int k, int lda, int ldb) {
  f.a[0] = ld_fragA(ap + k, lda);
  f.a[1] = ld_fragA(ap + (size_t)16 * lda + k, lda);
#pragma unroll
  for (int j = 0; j < 4; ++j)
    f.b[j] = ld_fragB(bp + (size_t)(j * 16) * ldb + k, ldb);
}

DEV void mma_frags(v8f acc[2][4], const Frags& f) {
#pragma unroll
  for (int i = 0; i < 2; ++i)
#pragma unroll
    for (int j = 0; j < 4; ++j)
      acc[i][j] = wmma_f16(f.a[i], f.b[j], acc[i][j]);
}

DEV void gemm_core(v8f acc[2][4], const _Float16* ap, const _Float16* bp,
                   int nIter, int lda, int ldb) {
  Frags E, O;
  load_frags(E, ap, bp, 0, lda, ldb);
  int it = 1;
  for (; it + 1 <= nIter - 1; it += 2) {
    load_frags(O, ap, bp, it * 32, lda, ldb);       // prefetch step it
    mma_frags(acc, E);                              // consume step it-1
    load_frags(E, ap, bp, (it + 1) * 32, lda, ldb); // prefetch step it+1
    mma_frags(acc, O);                              // consume step it
  }
  if (it <= nIter - 1) {   // one load left
    load_frags(O, ap, bp, it * 32, lda, ldb);
    mma_frags(acc, E);
    mma_frags(acc, O);
  } else {
    mma_frags(acc, E);
  }
}

DEV void zero_acc(v8f acc[2][4]) {
#pragma unroll
  for (int i = 0; i < 2; ++i)
#pragma unroll
    for (int j = 0; j < 4; ++j)
#pragma unroll
      for (int e = 0; e < 8; ++e) acc[i][j][e] = 0.0f;
}

// ---- conversion kernels ----------------------------------------------------

__global__ __launch_bounds__(256) void k_cvt(const float* __restrict__ in,
                                             _Float16* __restrict__ out, int n) {
  int i = blockIdx.x * 256 + threadIdx.x;
  if (i < n) out[i] = (_Float16)in[i];
}

// W [K x N] f32 -> Wt [N x K] f16 (so B-fragments are lane-contiguous).
__global__ __launch_bounds__(256) void k_transpose_cvt(const float* __restrict__ W,
                                                       _Float16* __restrict__ Wt) {
  __shared__ float tile[32][33];
  int n0 = blockIdx.x * 32, k0 = blockIdx.y * 32;
  int tx = threadIdx.x & 31, ty = threadIdx.x >> 5;  // 32 x 8
#pragma unroll
  for (int i = ty; i < 32; i += 8)
    tile[i][tx] = W[(size_t)(k0 + i) * Dm + (n0 + tx)];
  __syncthreads();
#pragma unroll
  for (int i = ty; i < 32; i += 8)
    Wt[(size_t)(n0 + i) * Dm + (k0 + tx)] = (_Float16)tile[tx][i];
}

// ---- generic WMMA GEMM:  C = A * Bt^T + bias -------------------------------
// Block: 8 waves as 4(m) x 2(n); block tile 128 x 128; wave tile 32 x 64.
// MODE 0: Cf f32 row-major     (final output projection)
// MODE 1: Ch f16 row-major     (Q, K)
// MODE 2: Ch f16 per-batch transposed [b][N][Sq]  (V -> Vt)
template <int MODE>
__global__ __launch_bounds__(256) void k_gemm(const _Float16* __restrict__ A,
                                              const _Float16* __restrict__ Bt,
                                              const float* __restrict__ bias,
                                              float* __restrict__ Cf,
                                              _Float16* __restrict__ Ch,
                                              int M, int N, int Kd) {
  int wave = threadIdx.x >> 5;
  int lane = threadIdx.x & 31;
  int wm = wave >> 1, wn = wave & 1;
  int m0 = blockIdx.y * 128 + wm * 32;
  int n0 = blockIdx.x * 128 + wn * 64;
  if (m0 >= M) return;

  v8f acc[2][4];
  zero_acc(acc);
  gemm_core(acc, A + (size_t)m0 * Kd, Bt + (size_t)n0 * Kd, Kd >> 5, Kd, Kd);

  int nlo = lane & 15, g = lane >> 4;
#pragma unroll
  for (int i = 0; i < 2; ++i) {
#pragma unroll
    for (int j = 0; j < 4; ++j) {
      int col = n0 + j * 16 + nlo;
      float bv = bias ? bias[col] : 0.0f;
#pragma unroll
      for (int r = 0; r < 8; ++r) {
        int m = m0 + i * 16 + g * 8 + r;
        float v = acc[i][j][r] + bv;
        if (MODE == 0) {
          Cf[(size_t)m * N + col] = v;
        } else if (MODE == 1) {
          Ch[(size_t)m * N + col] = (_Float16)v;
        } else {  // per-batch transposed: Vt[b][col][s]
          int bb = m >> 11, s = m & (Sq - 1);
          Ch[((size_t)bb * N + col) * Sq + s] = (_Float16)v;
        }
      }
    }
  }
}

// ---- causal scores: SP[b][q][k] = scale * (Q[b,q,:] . K[b,k,:]) ------------
// 32-query tiles: padded causal bound is exactly KL = q0 + 32.
// Masked/padding entries are written as -3e4 so softmax sees exp()->0.
__global__ __launch_bounds__(256) void k_scores(const _Float16* __restrict__ Qh,
                                                const _Float16* __restrict__ Kh,
                                                _Float16* __restrict__ SP,
                                                float scale) {
  int b = blockIdx.z;
  int wave = threadIdx.x >> 5, lane = threadIdx.x & 31;
  int q0 = blockIdx.y * 32;
  int k0 = blockIdx.x * 512 + wave * 64;
  int KL = q0 + 32;           // padded causal key bound for this q-tile
  if (k0 >= KL) return;

  v8f acc[2][4];
  zero_acc(acc);
  gemm_core(acc, Qh + ((size_t)b * Sq + q0) * Dm,
            Kh + ((size_t)b * Sq + k0) * Dm, Dm >> 5, Dm, Dm);

  int nlo = lane & 15, g = lane >> 4;
#pragma unroll
  for (int i = 0; i < 2; ++i) {
#pragma unroll
    for (int j = 0; j < 4; ++j) {
      int kc = k0 + j * 16 + nlo;
#pragma unroll
      for (int r = 0; r < 8; ++r) {
        int q = q0 + i * 16 + g * 8 + r;
        float v = (kc <= q) ? acc[i][j][r] * scale : -3.0e4f;
        SP[((size_t)b * Sq + q) * Sq + kc] = (_Float16)v;
      }
    }
  }
}

// ---- row softmax over the padded causal prefix, in place in f16 ------------
__global__ __launch_bounds__(256) void k_softmax(_Float16* __restrict__ SP) {
  int row = blockIdx.x;           // b*Sq + q
  int q = row & (Sq - 1);
  int L = ((q >> 5) + 1) * 32;    // same padded bound as k_scores/k_pv
  _Float16* p = SP + (size_t)row * Sq;
  __shared__ float red[8];
  int tid = threadIdx.x, lane = tid & 31, wid = tid >> 5;

  float m = -3.0e38f;
  for (int k = tid; k < L; k += 256) m = fmaxf(m, (float)p[k]);
#pragma unroll
  for (int off = 16; off > 0; off >>= 1) m = fmaxf(m, __shfl_xor(m, off, 32));
  if (lane == 0) red[wid] = m;
  __syncthreads();
  float mrow = red[0];
#pragma unroll
  for (int i = 1; i < 8; ++i) mrow = fmaxf(mrow, red[i]);
  __syncthreads();

  float s = 0.0f;
  for (int k = tid; k < L; k += 256) {
    float v = __expf((float)p[k] - mrow);
    s += v;
    p[k] = (_Float16)v;
  }
#pragma unroll
  for (int off = 16; off > 0; off >>= 1) s += __shfl_xor(s, off, 32);
  if (lane == 0) red[wid] = s;
  __syncthreads();
  float tot = 0.0f;
#pragma unroll
  for (int i = 0; i < 8; ++i) tot += red[i];
  float inv = 1.0f / tot;
  for (int k = tid; k < L; k += 256) p[k] = (_Float16)((float)p[k] * inv);
}

// ---- O[b][q][d] = P[b][q][:KL] * V[b][:KL][d], V pre-transposed ------------
__global__ __launch_bounds__(256) void k_pv(const _Float16* __restrict__ SP,
                                            const _Float16* __restrict__ Vt,
                                            _Float16* __restrict__ Oh) {
  int b = blockIdx.z;
  int wave = threadIdx.x >> 5, lane = threadIdx.x & 31;
  int q0 = blockIdx.y * 32;
  int n0 = blockIdx.x * 512 + wave * 64;
  int nIter = (q0 >> 5) + 1;   // KL/32 with KL = q0 + 32 (dynamic, may be odd)

  v8f acc[2][4];
  zero_acc(acc);
  gemm_core(acc, SP + ((size_t)b * Sq + q0) * Sq,
            Vt + ((size_t)b * Dm + n0) * Sq, nIter, Sq, Sq);

  int nlo = lane & 15, g = lane >> 4;
#pragma unroll
  for (int i = 0; i < 2; ++i) {
#pragma unroll
    for (int j = 0; j < 4; ++j) {
      int col = n0 + j * 16 + nlo;
#pragma unroll
      for (int r = 0; r < 8; ++r) {
        int q = q0 + i * 16 + g * 8 + r;
        Oh[((size_t)b * Sq + q) * Dm + col] = (_Float16)acc[i][j][r];
      }
    }
  }
}

// ---- host orchestration ----------------------------------------------------

extern "C" void kernel_launch(void* const* d_in, const int* in_sizes, int n_in,
                              void* d_out, int out_size, void* d_ws, size_t ws_size,
                              hipStream_t stream) {
  (void)in_sizes; (void)n_in; (void)out_size; (void)ws_size;
  const float* x  = (const float*)d_in[0];
  const float* Wq = (const float*)d_in[1];
  const float* bq = (const float*)d_in[2];
  const float* Wk = (const float*)d_in[3];
  const float* bk = (const float*)d_in[4];
  const float* Wv = (const float*)d_in[5];
  const float* bv = (const float*)d_in[6];
  const float* Wo = (const float*)d_in[7];
  const float* bo = (const float*)d_in[8];
  float* out = (float*)d_out;

  char* ws = (char*)d_ws;
  size_t off = 0;
  auto alloc = [&](size_t bytes) {
    char* p = ws + off;
    off += (bytes + 255) & ~(size_t)255;
    return p;
  };
  _Float16* Xh  = (_Float16*)alloc((size_t)NT * Dm * 2);      // 16 MB
  _Float16* Wqt = (_Float16*)alloc((size_t)Dm * Dm * 2);      // 2 MB
  _Float16* Wkt = (_Float16*)alloc((size_t)Dm * Dm * 2);
  _Float16* Wvt = (_Float16*)alloc((size_t)Dm * Dm * 2);
  _Float16* Wot = (_Float16*)alloc((size_t)Dm * Dm * 2);
  _Float16* Qh  = (_Float16*)alloc((size_t)NT * Dm * 2);      // 16 MB
  _Float16* Kh  = (_Float16*)alloc((size_t)NT * Dm * 2);      // 16 MB
  _Float16* Vth = (_Float16*)alloc((size_t)NT * Dm * 2);      // 16 MB (transposed)
  _Float16* SP  = (_Float16*)alloc((size_t)Bn * Sq * Sq * 2); // 33.5 MB
  _Float16* Oh  = (_Float16*)alloc((size_t)NT * Dm * 2);      // 16 MB
  // total ~122 MB of workspace

  dim3 blk(256);

  // 1) fp32 -> fp16 conversions (+ weight transposes)
  k_cvt<<<dim3((NT * Dm + 255) / 256), blk, 0, stream>>>(x, Xh, NT * Dm);
  dim3 tg(Dm / 32, Dm / 32);
  k_transpose_cvt<<<tg, blk, 0, stream>>>(Wq, Wqt);
  k_transpose_cvt<<<tg, blk, 0, stream>>>(Wk, Wkt);
  k_transpose_cvt<<<tg, blk, 0, stream>>>(Wv, Wvt);
  k_transpose_cvt<<<tg, blk, 0, stream>>>(Wo, Wot);

  // 2) QKV projections (WMMA GEMMs); V written transposed for P*V
  dim3 gg(Dm / 128, NT / 128);
  k_gemm<1><<<gg, blk, 0, stream>>>(Xh, Wqt, bq, nullptr, Qh, NT, Dm, Dm);
  k_gemm<1><<<gg, blk, 0, stream>>>(Xh, Wkt, bk, nullptr, Kh, NT, Dm, Dm);
  k_gemm<2><<<gg, blk, 0, stream>>>(Xh, Wvt, bv, nullptr, Vth, NT, Dm, Dm);

  // 3) causal scores (scale = 1/sqrt(1024) = 1/32)
  k_scores<<<dim3(Sq / 512, Sq / 32, Bn), blk, 0, stream>>>(Qh, Kh, SP, 1.0f / 32.0f);

  // 4) row softmax in place
  k_softmax<<<dim3(NT), blk, 0, stream>>>(SP);

  // 5) attention-weighted values
  k_pv<<<dim3(Dm / 512, Sq / 32, Bn), blk, 0, stream>>>(SP, Vth, Oh);

  // 6) output projection -> fp32 d_out
  k_gemm<0><<<gg, blk, 0, stream>>>(Oh, Wot, bo, out, nullptr, NT, Dm, Dm);
}
